// ProgammedController_83485574299999
// MI455X (gfx1250) — matrix operand order repaired and checked by reading, not compiled, not run on hardware
//
#include <hip/hip_runtime.h>
#include <hip/hip_bf16.h>

typedef __attribute__((ext_vector_type(16))) _Float16 v16h;
typedef __attribute__((ext_vector_type(8)))  _Float16 v8h;
typedef __attribute__((ext_vector_type(8)))  float    v8f;

#define DELTA_F 1e-6f

__device__ __forceinline__ v8f wmma16(v16h a, v16h b, v8f c) {
  return __builtin_amdgcn_wmma_f32_16x16x32_f16(false, a, false, b, (short)0, c,
                                                false, false);
}

// CDNA5 async global->LDS copy: 64 contiguous bytes from gsrc into LDS at
// ldsb. INST_OFFSET is added to BOTH the global and LDS addresses (ISA
// 15.18.3), so one base + immediate offsets covers the whole span. Tracked on
// ASYNCcnt; caller must s_wait_asynccnt + barrier before reading LDS.
__device__ __forceinline__ void async_copy64B(unsigned ldsb, const void* gsrc) {
  asm volatile("global_load_async_to_lds_b128 %0, %1, off"
               :: "v"(ldsb), "v"(gsrc) : "memory");
  asm volatile("global_load_async_to_lds_b128 %0, %1, off offset:16"
               :: "v"(ldsb), "v"(gsrc) : "memory");
  asm volatile("global_load_async_to_lds_b128 %0, %1, off offset:32"
               :: "v"(ldsb), "v"(gsrc) : "memory");
  asm volatile("global_load_async_to_lds_b128 %0, %1, off offset:48"
               :: "v"(ldsb), "v"(gsrc) : "memory");
}

__device__ __forceinline__ void wait_async0() {
  asm volatile("s_wait_asynccnt 0" ::: "memory");
}

// A-matrix 16x32 f16 fragment from an f32 row pointer (contiguous K).
// Element e -> K = k0 + 16*hi + e
__device__ __forceinline__ v16h load_a_frag_f32(const float* __restrict__ rowp,
                                                int k0, int hi) {
  const float4* p0 = reinterpret_cast<const float4*>(rowp + k0 + 16 * hi);
  const float4* p1 = reinterpret_cast<const float4*>(rowp + k0 + 16 * hi + 8);
  float4 u0 = p0[0], u1 = p0[1], u2 = p1[0], u3 = p1[1];
  v16h a;
  a[0]  = (_Float16)u0.x; a[1]  = (_Float16)u0.y; a[2]  = (_Float16)u0.z; a[3]  = (_Float16)u0.w;
  a[4]  = (_Float16)u1.x; a[5]  = (_Float16)u1.y; a[6]  = (_Float16)u1.z; a[7]  = (_Float16)u1.w;
  a[8]  = (_Float16)u2.x; a[9]  = (_Float16)u2.y; a[10] = (_Float16)u2.z; a[11] = (_Float16)u2.w;
  a[12] = (_Float16)u3.x; a[13] = (_Float16)u3.y; a[14] = (_Float16)u3.z; a[15] = (_Float16)u3.w;
  return a;
}

// B-matrix 32x16 f16 fragment: element e -> K = 16*hi + e ; p points at the
// 16 contiguous f32 K-values this lane needs (aligned version).
__device__ __forceinline__ v16h load_b_frag_f32_vec(const float* __restrict__ p) {
  const float4* p4 = reinterpret_cast<const float4*>(p);
  float4 u0 = p4[0], u1 = p4[1], u2 = p4[2], u3 = p4[3];
  v16h b;
  b[0]  = (_Float16)u0.x; b[1]  = (_Float16)u0.y; b[2]  = (_Float16)u0.z; b[3]  = (_Float16)u0.w;
  b[4]  = (_Float16)u1.x; b[5]  = (_Float16)u1.y; b[6]  = (_Float16)u1.z; b[7]  = (_Float16)u1.w;
  b[8]  = (_Float16)u2.x; b[9]  = (_Float16)u2.y; b[10] = (_Float16)u2.z; b[11] = (_Float16)u2.w;
  b[12] = (_Float16)u3.x; b[13] = (_Float16)u3.y; b[14] = (_Float16)u3.z; b[15] = (_Float16)u3.w;
  return b;
}

// Scalar (possibly-unaligned) version for PM_S (row stride 65 floats).
__device__ __forceinline__ v16h load_b_frag_f32_scalar(const float* __restrict__ p) {
  v16h b;
#pragma unroll
  for (int e = 0; e < 16; e++) b[e] = (_Float16)p[e];
  return b;
}

// ---------------------------------------------------------------------------
// Stage 1: k_c = x @ W_c^T + b_c     (M=64, N=4096, K=1024)
// One wave per 16x16 output tile. grid = (256 nt, 4 mt), block = 32.
// ---------------------------------------------------------------------------
__global__ void k_proj(const float* __restrict__ x, const float* __restrict__ W,
                       const float* __restrict__ bias, float* __restrict__ kout) {
  int lane = threadIdx.x, lo = lane & 15, hi = lane >> 4;
  int nt = blockIdx.x, mt = blockIdx.y;
  const float* xr = x + (size_t)(mt * 16 + lo) * 1024;
  const float* wr = W + (size_t)(nt * 16 + lo) * 1024;
  v8f acc = {};
  for (int k0 = 0; k0 < 1024; k0 += 32) {
    v16h a = load_a_frag_f32(xr, k0, hi);
    v16h b = load_b_frag_f32_vec(wr + k0 + 16 * hi);
    acc = wmma16(a, b, acc);
  }
  int n = nt * 16 + lo;
  float bb = bias[n];
#pragma unroll
  for (int r = 0; r < 8; r++) {
    int m = mt * 16 + r + 8 * hi;
    kout[m * 4096 + n] = acc[r] + bb;
  }
}

// ---------------------------------------------------------------------------
// Stage 2: num[b,f,p] = sum_k k[b,f,k] * MK[p,k]   (per-batch 64x128x64 GEMM)
// grid = (8 nt, 4 mt, 64 b), block = 32. pmStride in floats (2112 or 65).
// ---------------------------------------------------------------------------
__global__ void k_theta(const float* __restrict__ kbuf, const float* __restrict__ PM,
                        int pmStride, float* __restrict__ numout) {
  int lane = threadIdx.x, lo = lane & 15, hi = lane >> 4;
  int nt = blockIdx.x, mt = blockIdx.y, b = blockIdx.z;
  int f = mt * 16 + lo;
  int p = nt * 16 + lo;
  const float* ar = kbuf + (b << 12) + (f << 6);
  const float* br = PM + (size_t)p * pmStride;
  v8f acc = {};
#pragma unroll
  for (int k0 = 0; k0 < 64; k0 += 32) {
    v16h a = load_a_frag_f32(ar, k0, hi);
    v16h bf = load_b_frag_f32_scalar(br + k0 + 16 * hi);
    acc = wmma16(a, bf, acc);
  }
#pragma unroll
  for (int r = 0; r < 8; r++) {
    int m = mt * 16 + r + 8 * hi;
    numout[(b << 13) + m * 128 + p] = acc[r];
  }
}

// an[p] = ||MK_p|| + DELTA. grid=1, block=128.
__global__ void k_an(const float* __restrict__ PM, int pmStride, float* __restrict__ an) {
  int p = threadIdx.x;
  const float* r = PM + (size_t)p * pmStride;
  float s = 0.f;
  for (int k = 0; k < 64; k++) { float v = r[k]; s += v * v; }
  an[p] = sqrtf(s) + DELTA_F;
}

// ---------------------------------------------------------------------------
// Stage 3: d = softmax_p( num / (64 * bn[b,f] * an[p] + DELTA) ), in place.
// grid = (64 f, 64 b), block = 128 (one row of P=128).
// ---------------------------------------------------------------------------
__global__ void k_softmax(const float* __restrict__ kbuf, const float* __restrict__ an,
                          float* __restrict__ dbuf) {
  int f = blockIdx.x, b = blockIdx.y, t = threadIdx.x;
  __shared__ float red[128];
  float kv = 0.f;
  if (t < 64) { float q = kbuf[(b << 12) + (f << 6) + t]; kv = q * q; }
  red[t] = kv;
  __syncthreads();
  for (int off = 64; off >= 1; off >>= 1) {
    if (t < off) red[t] += red[t + off];
    __syncthreads();
  }
  float bn = sqrtf(red[0]) + DELTA_F;
  __syncthreads();
  int idx = (b << 13) + (f << 7) + t;
  float num = dbuf[idx];
  float logit = num / (64.f * bn * an[t] + DELTA_F);
  red[t] = logit;
  __syncthreads();
  for (int off = 64; off >= 1; off >>= 1) {
    if (t < off) red[t] = fmaxf(red[t], red[t + off]);
    __syncthreads();
  }
  float mx = red[0];
  __syncthreads();
  float e = __expf(logit - mx);
  red[t] = e;
  __syncthreads();
  for (int off = 64; off >= 1; off >>= 1) {
    if (t < off) red[t] += red[t + off];
    __syncthreads();
  }
  dbuf[idx] = e / red[0];
}

// ---------------------------------------------------------------------------
// Stage 4: Us/Vs[b,f,n] = sum_p d[b,f,p] * MP[p,n]  (per-batch 64x2048x128),
// output stored as f16. grid = (16 nblk, 64 b), block = 256 (8 waves).
// B tile (32 x 128) staged through LDS each K-step. PM stride fixed = 2112.
// ---------------------------------------------------------------------------
__global__ void k_read(const float* __restrict__ dbuf, const float* __restrict__ PM,
                       _Float16* __restrict__ outH) {
  __shared__ __align__(16) _Float16 Bl[32 * 128];
  int t = threadIdx.x;
  int n0 = blockIdx.x * 128;
  int b = blockIdx.y;
  int w = t >> 5, lane = t & 31, lo = lane & 15, hi = lane >> 4;
  v8f acc[4] = {{}, {}, {}, {}};
  for (int k0 = 0; k0 < 128; k0 += 32) {
    // cooperative stage of MP[k0:k0+32, n0:n0+128] as f16
    {
      int kkl = t >> 3;          // 0..31
      int seg = (t & 7) * 16;    // 0..112
      const float4* src = reinterpret_cast<const float4*>(
          PM + (size_t)(k0 + kkl) * 2112 + 64 + n0 + seg);
      _Float16* dst = &Bl[kkl * 128 + seg];
#pragma unroll
      for (int q = 0; q < 4; q++) {
        float4 v = src[q];
        dst[q * 4 + 0] = (_Float16)v.x;
        dst[q * 4 + 1] = (_Float16)v.y;
        dst[q * 4 + 2] = (_Float16)v.z;
        dst[q * 4 + 3] = (_Float16)v.w;
      }
    }
    __syncthreads();
    v16h bfr;
#pragma unroll
    for (int e = 0; e < 16; e++) bfr[e] = Bl[(16 * hi + e) * 128 + w * 16 + lo];
#pragma unroll
    for (int mt = 0; mt < 4; mt++) {
      const float* ar = dbuf + (((b << 6) + mt * 16 + lo) << 7);
      v16h a = load_a_frag_f32(ar, k0, hi);
      acc[mt] = wmma16(a, bfr, acc[mt]);
    }
    __syncthreads();
  }
#pragma unroll
  for (int mt = 0; mt < 4; mt++)
#pragma unroll
    for (int r = 0; r < 8; r++) {
      int f = mt * 16 + r + 8 * hi;
      int n = n0 + w * 16 + lo;
      outH[((size_t)(b << 6) + f) * 2048 + n] = (_Float16)acc[mt][r];
    }
}

// ---------------------------------------------------------------------------
// Stage 5: s[b,f] = cumsum_f softplus( sum_p dS[b,f,p] * MP_S[p] ).
// grid = 64 (b), block = 64 (f).
// ---------------------------------------------------------------------------
__global__ void k_scale(const float* __restrict__ dS, const float* __restrict__ PMS,
                        float* __restrict__ sOut) {
  int b = blockIdx.x, f = threadIdx.x;
  __shared__ float sh[64];
  const float* dr = dS + (((b << 6) + f) << 7);
  float acc = 0.f;
  for (int p = 0; p < 128; p++) acc += dr[p] * PMS[p * 65 + 64];
  float sp = (acc > 20.f) ? acc : log1pf(__expf(acc));
  sh[f] = sp;
  __syncthreads();
  for (int off = 1; off < 64; off <<= 1) {
    float add = (f >= off) ? sh[f - off] : 0.f;
    __syncthreads();
    sh[f] += add;
    __syncthreads();
  }
  sOut[(b << 6) + f] = sh[f];
}

// ---------------------------------------------------------------------------
// Stage 6: Wrec[b,i,j] = sum_f (s[b,f]*Us[b,f,i]) * Vs[b,f,j]
// Per-batch GEMM M=N=2048, K=64. Workgroup = 128x128 output block, 8 waves,
// wave w owns i-tile w and loops 8 j-tiles. A (scaled) staged manually; B
// staged via CDNA5 async global->LDS (ASYNCcnt). Output stores are
// non-temporal (1 GiB stream written once, never re-read) so L2 keeps the
// 33 MB Us/Vs f16 working set hot across the 16x16 grid of output blocks.
// grid = (16 jblk, 16 iblk, 64 b), block = 256.
// ---------------------------------------------------------------------------
__global__ void k_wrec(const _Float16* __restrict__ UsH, const _Float16* __restrict__ VsH,
                       const float* __restrict__ sBuf, float* __restrict__ out) {
  __shared__ __align__(16) _Float16 Al[64 * 128];
  __shared__ __align__(16) _Float16 Bl[64 * 128];
  int t = threadIdx.x;
  int b = blockIdx.z;
  int i0 = blockIdx.y * 128, j0 = blockIdx.x * 128;
  {
    int f = t >> 2;
    int seg = (t & 3) * 32;
    // B: straight 16 KB copy -> async global->LDS, overlapped with the A cvt
    const _Float16* vsrc = VsH + ((size_t)(b << 6) + f) * 2048 + j0 + seg;
    _Float16* bd = &Bl[f * 128 + seg];
    async_copy64B((unsigned)(unsigned long long)(void*)bd, (const void*)vsrc);

    // A: manual stage (applies the diag(s) scale during the f16 write)
    float sf = sBuf[(b << 6) + f];
    const v8h* ua = reinterpret_cast<const v8h*>(
        UsH + ((size_t)(b << 6) + f) * 2048 + i0 + seg);
    _Float16* ad = &Al[f * 128 + seg];
#pragma unroll
    for (int q = 0; q < 4; q++) {
      v8h hu = ua[q];
#pragma unroll
      for (int e = 0; e < 8; e++) ad[q * 8 + e] = (_Float16)((float)hu[e] * sf);
    }
    wait_async0();
  }
  __syncthreads();
  int w = t >> 5, lane = t & 31, lo = lane & 15, hi = lane >> 4;
  // A fragments for both K-steps, reused across all 8 j-tiles.
  v16h afr[2];
#pragma unroll
  for (int ks = 0; ks < 2; ks++)
#pragma unroll
    for (int e = 0; e < 16; e++) {
      int f = ks * 32 + 16 * hi + e;
      afr[ks][e] = Al[f * 128 + w * 16 + lo];
    }
#pragma unroll
  for (int jt = 0; jt < 8; jt++) {
    v8f acc = {};
#pragma unroll
    for (int ks = 0; ks < 2; ks++) {
      v16h bfr;
#pragma unroll
      for (int e = 0; e < 16; e++)
        bfr[e] = Bl[(ks * 32 + 16 * hi + e) * 128 + jt * 16 + lo];
      acc = wmma16(afr[ks], bfr, acc);
    }
#pragma unroll
    for (int r = 0; r < 8; r++) {
      size_t i = (size_t)(i0 + w * 16 + r + 8 * hi);
      size_t j = (size_t)(j0 + jt * 16 + lo);
      __builtin_nontemporal_store(acc[r], &out[((size_t)b << 22) + i * 2048 + j]);
    }
  }
}

extern "C" void kernel_launch(void* const* d_in, const int* in_sizes, int n_in,
                              void* d_out, int out_size, void* d_ws, size_t ws_size,
                              hipStream_t stream) {
  (void)in_sizes; (void)n_in; (void)out_size; (void)ws_size;
  const float* x   = (const float*)d_in[0];
  const float* PMU = (const float*)d_in[1];
  const float* PMV = (const float*)d_in[2];
  const float* PMS = (const float*)d_in[3];
  const float* Wku = (const float*)d_in[4];
  const float* bku = (const float*)d_in[5];
  const float* Wkv = (const float*)d_in[6];
  const float* bkv = (const float*)d_in[7];
  const float* Wks = (const float*)d_in[8];
  const float* bks = (const float*)d_in[9];
  float* out = (float*)d_out;

  // workspace layout (floats then halves), ~43 MB total
  float* ws  = (float*)d_ws;
  float* kU  = ws;                 // 64*4096
  float* kV  = kU + 262144;
  float* kS  = kV + 262144;
  float* dU  = kS + 262144;        // 64*64*128
  float* dV  = dU + 524288;
  float* dS  = dV + 524288;
  float* anU = dS + 524288;        // 128 each
  float* anV = anU + 128;
  float* anS = anV + 128;
  float* sB  = anS + 128;          // 64*64 (padded region start)
  _Float16* UsH = (_Float16*)(sB + 4096);   // 64*64*2048 halves
  _Float16* VsH = UsH + (size_t)64 * 64 * 2048;

  dim3 gProj(256, 4);
  k_proj<<<gProj, 32, 0, stream>>>(x, Wku, bku, kU);
  k_proj<<<gProj, 32, 0, stream>>>(x, Wkv, bkv, kV);
  k_proj<<<gProj, 32, 0, stream>>>(x, Wks, bks, kS);

  dim3 gTheta(8, 4, 64);
  k_theta<<<gTheta, 32, 0, stream>>>(kU, PMU, 2112, dU);
  k_theta<<<gTheta, 32, 0, stream>>>(kV, PMV, 2112, dV);
  k_theta<<<gTheta, 32, 0, stream>>>(kS, PMS, 65, dS);

  k_an<<<1, 128, 0, stream>>>(PMU, 2112, anU);
  k_an<<<1, 128, 0, stream>>>(PMV, 2112, anV);
  k_an<<<1, 128, 0, stream>>>(PMS, 65, anS);

  dim3 gSm(64, 64);
  k_softmax<<<gSm, 128, 0, stream>>>(kU, anU, dU);
  k_softmax<<<gSm, 128, 0, stream>>>(kV, anV, dV);
  k_softmax<<<gSm, 128, 0, stream>>>(kS, anS, dS);

  dim3 gRd(16, 64);
  k_read<<<gRd, 256, 0, stream>>>(dU, PMU, UsH);
  k_read<<<gRd, 256, 0, stream>>>(dV, PMV, VsH);

  k_scale<<<64, 64, 0, stream>>>(dS, PMS, sB);

  dim3 gW(16, 16, 64);
  k_wrec<<<gW, 256, 0, stream>>>(UsH, VsH, sB, out);
}
